// Trans4Map_3195455668800
// MI455X (gfx1250) — matrix operand order repaired and checked by reading, not compiled, not run on hardware
//
#include <hip/hip_runtime.h>
#include <hip/hip_bf16.h>
#include <math.h>

// ---------------- problem constants (match reference) ----------------
#define T_STEPS 10
#define EGO_C   64
#define FH_     120
#define FW_     160
#define OH_     480
#define OW_     640
#define NPIX    (OH_*OW_)        // 307200
#define MEM_C_  128
#define MAPH    200
#define MAPW    200
#define MSIZE   (MAPH*MAPW)      // 40000
#define GATE3   (3*MEM_C_)       // 384
#define SEM_N   21

typedef __attribute__((ext_vector_type(16))) _Float16 v16h;
typedef __attribute__((ext_vector_type(8)))  _Float16 v8h;
typedef __attribute__((ext_vector_type(8)))  float    v8f;

// ---------------- helpers ----------------
__device__ __forceinline__ unsigned orderkey(float f) {
  unsigned u = __float_as_uint(f);
  return (u & 0x80000000u) ? ~u : (u | 0x80000000u);
}
__device__ __forceinline__ float fromkey(unsigned k) {
  unsigned u = (k & 0x80000000u) ? (k & 0x7FFFFFFFu) : ~k;
  return __uint_as_float(u);
}

// ---------------- generic utility kernels ----------------
__global__ void k_zero32(unsigned* p, int n) {
  int i = blockIdx.x * blockDim.x + threadIdx.x;
  if (i < n) p[i] = 0u;
}

__global__ void k_f32_to_f16(const float* __restrict__ s, _Float16* __restrict__ d, int n) {
  int i = blockIdx.x * blockDim.x + threadIdx.x;
  if (i < n) d[i] = (_Float16)s[i];
}

// Repack conv weights OIHW f32 -> [Opad][KH*KW*Cpad] f16 (zero padding)
__global__ void k_pack_conv(const float* __restrict__ src, _Float16* __restrict__ dst,
                            int O, int Opad, int I, int Cpad, int KH, int KW) {
  int Ktot = KH * KW * Cpad;
  int n = Opad * Ktot;
  int idx = blockIdx.x * blockDim.x + threadIdx.x;
  if (idx >= n) return;
  int o = idx / Ktot;
  int kk = idx - o * Ktot;
  int tap = kk / Cpad;
  int c = kk - tap * Cpad;
  float v = 0.0f;
  if (o < O && c < I) {
    int ky = tap / KW, kx = tap - (tap / KW) * KW;
    v = src[((size_t)o * I + c) * (KH * KW) + (size_t)ky * KW + kx];
  }
  dst[idx] = (_Float16)v;
}

// ---------------- scatter phase ----------------
__global__ void k_cell_prep(const float* __restrict__ hm, unsigned* __restrict__ keyb,
                            int* __restrict__ argb) {
  int i = blockIdx.x * blockDim.x + threadIdx.x;
  if (i < MSIZE) { keyb[i] = orderkey(hm[i]); argb[i] = -1; }
}

__global__ void k_scatter_max(const int* __restrict__ proj, const unsigned char* __restrict__ mask,
                              const float* __restrict__ hgt, unsigned* __restrict__ keyb) {
  int p = blockIdx.x * blockDim.x + threadIdx.x;
  if (p >= NPIX) return;
  int cx = proj[2 * p], cy = proj[2 * p + 1];
  int cell = MAPW * cy + cx;
  cell = min(max(cell, 0), MSIZE - 1);
  float hf = (mask[p] == 0) ? (hgt[p] + 1000.0f) : -INFINITY;
  atomicMax(&keyb[cell], orderkey(hf));
}

__global__ void k_scatter_arg(const int* __restrict__ proj, const unsigned char* __restrict__ mask,
                              const float* __restrict__ hgt, const float* __restrict__ hm_prev,
                              const unsigned* __restrict__ keyb, int* __restrict__ argb) {
  int p = blockIdx.x * blockDim.x + threadIdx.x;
  if (p >= NPIX) return;
  if (mask[p] != 0) return;                       // outlier -> -inf, never wins
  int cx = proj[2 * p], cy = proj[2 * p + 1];
  int cell = MAPW * cy + cx;
  cell = min(max(cell, 0), MSIZE - 1);
  float hf = hgt[p] + 1000.0f;
  if (hf > hm_prev[cell] && orderkey(hf) == keyb[cell])
    atomicMax(&argb[cell], p);
}

// finalize cell max into hm, mark obs, and gather bilinear (align_corners) ego
// features of the winning pixel into f16 GRU input matrix G [MSIZE][EGO_C]
__global__ void k_finalize_gather(const unsigned* __restrict__ keyb, const int* __restrict__ argb,
                                  float* __restrict__ hm, unsigned* __restrict__ obs, int write_obs,
                                  const float* __restrict__ feat, _Float16* __restrict__ G) {
  int t = threadIdx.x;
  int cell = blockIdx.x * 4 + (t >> 6);
  int c = t & 63;
  if (cell >= MSIZE) return;
  int a = argb[cell];
  if (c == 0) {
    hm[cell] = fromkey(keyb[cell]);
    if (write_obs && a >= 0) obs[cell] = 1u;
  }
  float val = 0.0f;
  if (a >= 0) {
    int y = a / OW_, x = a - (a / OW_) * OW_;
    float ys = y * ((float)(FH_ - 1) / (float)(OH_ - 1));
    float xs = x * ((float)(FW_ - 1) / (float)(OW_ - 1));
    int y0 = (int)floorf(ys); int y1 = min(y0 + 1, FH_ - 1); float wy = ys - (float)y0;
    int x0 = (int)floorf(xs); int x1 = min(x0 + 1, FW_ - 1); float wx = xs - (float)x0;
    const float* fc = feat + (size_t)c * FH_ * FW_;
    float g0 = fc[y0 * FW_ + x0] * (1.0f - wy) + fc[y1 * FW_ + x0] * wy;
    float g1 = fc[y0 * FW_ + x1] * (1.0f - wy) + fc[y1 * FW_ + x1] * wy;
    val = g0 * (1.0f - wx) + g1 * wx;
  }
  G[(size_t)cell * EGO_C + c] = (_Float16)val;
}

// ---------------- unified WMMA GEMM / implicit-im2col conv ----------------
// out[M][NOUT] = act(im2col)[M][KH*KW*Cin] @ wgt[NOUT][KH*KW*Cin]^T
// act pixel-major [H*W][Cin] f16. All shape params are compile-time so every
// division/modulo strength-reduces. One wave owns a (16*MT) x (16*NT) C tile.
// C/D layout per CDNA5 ISA 7.12.2 (wave32); A/B use contiguous K (k = 16h + s).
template <int NT, int MT, int H, int W, int Cin, int KH, int KW, int PAD, int NOUT>
__global__ __launch_bounds__(128) void k_conv_wmma(
    const _Float16* __restrict__ act, const _Float16* __restrict__ wgt,
    float* __restrict__ out) {
  constexpr int Ktot = KH * KW * Cin;
  constexpr int M = H * W;
  constexpr int mgroups = M / (16 * MT);
  constexpr int ngroups = NOUT / (16 * NT);
  const int wave = blockIdx.x * (blockDim.x >> 5) + (threadIdx.x >> 5);
  if (wave >= mgroups * ngroups) return;
  const int mt = wave / ngroups;
  const int ng = wave - mt * ngroups;
  const int m0 = mt * (16 * MT);
  const int n0 = ng * (16 * NT);
  const int l = threadIdx.x & 31;
  const int r16 = l & 15;
  const int grp = l >> 4;      // 0: lanes 0-15, 1: lanes 16-31

  int oy[MT], ox[MT];
#pragma unroll
  for (int i = 0; i < MT; ++i) {
    int m = m0 + i * 16 + r16;      // A row held by this lane (both lane groups)
    oy[i] = m / W;
    ox[i] = m - oy[i] * W;
  }

  v8f acc[MT][NT] = {};
#pragma unroll 2
  for (int k0 = 0; k0 < Ktot; k0 += 32) {
    const int tap = k0 / Cin;          // K chunk never crosses a tap (Cin%32==0)
    const int c0 = k0 - tap * Cin;
    const int ky = tap / KW;
    const int kx = tap - ky * KW;
    // A 16x32 f16 fragments: lanes 0-15 K c0+0..15; lanes 16-31 K c0+16..31
    v16h a[MT];
#pragma unroll
    for (int i = 0; i < MT; ++i) {
      const int iy = oy[i] + ky - PAD;
      const int ix = ox[i] + kx - PAD;
      a[i] = v16h{};
      if ((unsigned)iy < (unsigned)H && (unsigned)ix < (unsigned)W) {
        const _Float16* ab = act + (size_t)(iy * W + ix) * Cin + c0 + grp * 16;
        v8h lo = *(const v8h*)ab;
        v8h hi = *(const v8h*)(ab + 8);
        a[i] = __builtin_shufflevector(lo, hi, 0, 1, 2, 3, 4, 5, 6, 7,
                                       8, 9, 10, 11, 12, 13, 14, 15);
      }
    }
    if (k0 + 32 < Ktot)
      __builtin_prefetch(wgt + (size_t)(n0 + r16) * Ktot + k0 + 32, 0, 3);
#pragma unroll
    for (int j = 0; j < NT; ++j) {
      // B 32x16 f16 fragment: lane -> column n, lane group selects K 0-15/16-31
      const _Float16* bb = wgt + (size_t)(n0 + j * 16 + r16) * Ktot + k0 + grp * 16;
      v16h b = *(const v16h*)bb;
#pragma unroll
      for (int i = 0; i < MT; ++i)
        acc[i][j] = __builtin_amdgcn_wmma_f32_16x16x32_f16(false, a[i], false, b,
                                                           (short)0, acc[i][j], false, false);
    }
  }
  // C/D 16x16 f32: VGPR r -> row r (+8 for lanes 16-31), lane -> column
#pragma unroll
  for (int i = 0; i < MT; ++i) {
    const int mb = m0 + i * 16 + grp * 8;
#pragma unroll
    for (int j = 0; j < NT; ++j) {
      float* op = out + (size_t)mb * NOUT + n0 + j * 16 + r16;
#pragma unroll
      for (int rr = 0; rr < 8; ++rr) op[(size_t)rr * NOUT] = acc[i][j][rr];
    }
  }
}

template <int NT, int MT, int H, int W, int Cin, int KH, int KW, int PAD, int NOUT>
static inline void launch_conv(const _Float16* a, const _Float16* w, float* o, hipStream_t s) {
  constexpr int waves = (H * W / (16 * MT)) * (NOUT / (16 * NT));
  k_conv_wmma<NT, MT, H, W, Cin, KH, KW, PAD, NOUT><<<(waves + 3) / 4, 128, 0, s>>>(a, w, o);
}

// ---------------- GRU pointwise gates + masked state update ----------------
// Also maintains a persistent f16 shadow of the state for the next step's GEMM.
__global__ void k_gru_point(const float* __restrict__ gi, const float* __restrict__ gh,
                            const float* __restrict__ bih, const float* __restrict__ bhh,
                            const int* __restrict__ argb, float* __restrict__ state,
                            _Float16* __restrict__ stateH) {
  int idx = blockIdx.x * blockDim.x + threadIdx.x;
  if (idx >= MSIZE * MEM_C_) return;
  int cell = idx >> 7;
  int j = idx & (MEM_C_ - 1);
  if (argb[cell] < 0) return;            // jnp.where(m, ns, state)
  const float* gic = gi + (size_t)cell * GATE3;
  const float* ghc = gh + (size_t)cell * GATE3;
  float ir = gic[j] + bih[j],                 hr = ghc[j] + bhh[j];
  float iz = gic[128 + j] + bih[128 + j],     hz = ghc[128 + j] + bhh[128 + j];
  float in_ = gic[256 + j] + bih[256 + j],    hn = ghc[256 + j] + bhh[256 + j];
  float r = 1.0f / (1.0f + expf(-(ir + hr)));
  float z = 1.0f / (1.0f + expf(-(iz + hz)));
  float n = tanhf(in_ + r * hn);
  float h = state[idx];
  float ns = (1.0f - z) * n + z * h;
  state[idx] = ns;
  stateH[idx] = (_Float16)ns;
}

// ---------------- epilogues ----------------
__global__ void k_concat_half(const float* __restrict__ s, const float* __restrict__ sr,
                              _Float16* __restrict__ X) {
  int idx = blockIdx.x * blockDim.x + threadIdx.x;
  if (idx >= MSIZE * 256) return;
  int pix = idx >> 8, c = idx & 255;
  float v = (c < MEM_C_) ? s[(size_t)pix * MEM_C_ + c] : sr[(size_t)pix * MEM_C_ + c - MEM_C_];
  X[idx] = (_Float16)v;
}

__global__ void k_bias_half(const float* __restrict__ x, const float* __restrict__ bias,
                            _Float16* __restrict__ out, int Nc) {
  int idx = blockIdx.x * blockDim.x + threadIdx.x;
  if (idx >= MSIZE * Nc) return;
  int c = idx % Nc;
  out[idx] = (_Float16)(x[idx] + bias[c]);
}

__global__ void k_bnrelu(const float* __restrict__ x, int Nc, _Float16* __restrict__ out, int Cpad,
                         const float* __restrict__ g, const float* __restrict__ b,
                         const float* __restrict__ mm, const float* __restrict__ vv) {
  int idx = blockIdx.x * blockDim.x + threadIdx.x;
  if (idx >= MSIZE * Cpad) return;
  int pix = idx / Cpad, c = idx - (idx / Cpad) * Cpad;
  float y = 0.0f;
  if (c < Nc) {
    float xv = x[(size_t)pix * Nc + c];
    float inv = rsqrtf(vv[c] + 1e-5f);
    y = (xv - mm[c]) * inv * g[c] + b[c];
    y = y > 0.0f ? y : 0.0f;
  }
  out[idx] = (_Float16)y;
}

__global__ void k_c5_out(const float* __restrict__ x /*[MSIZE][32]*/, const float* __restrict__ b,
                         float* __restrict__ out) {
  int idx = blockIdx.x * blockDim.x + threadIdx.x;
  if (idx >= SEM_N * MSIZE) return;
  int n = idx / MSIZE, pix = idx - n * MSIZE;
  out[idx] = x[(size_t)pix * 32 + n] + b[n];
}

__global__ void k_out_obs_hm(const unsigned* __restrict__ obs, const float* __restrict__ hm,
                             float* __restrict__ out) {
  int i = blockIdx.x * blockDim.x + threadIdx.x;
  if (i >= MSIZE) return;
  out[SEM_N * MSIZE + i] = obs[i] ? 1.0f : 0.0f;
  out[SEM_N * MSIZE + MSIZE + i] = hm[i];
}

// ---------------- host orchestration ----------------
static inline int cdiv(int a, int b) { return (a + b - 1) / b; }

extern "C" void kernel_launch(void* const* d_in, const int* in_sizes, int n_in,
                              void* d_out, int out_size, void* d_ws, size_t ws_size,
                              hipStream_t stream) {
  (void)in_sizes; (void)n_in; (void)out_size; (void)ws_size;
  // input order: features, heights, params(leaf order), proj_wtm, mask, map_h, map_w
  const float* features = (const float*)d_in[0];
  const float* heights  = (const float*)d_in[1];
  const float* Wih[2] = {(const float*)d_in[2], (const float*)d_in[6]};
  const float* Whh[2] = {(const float*)d_in[3], (const float*)d_in[7]};
  const float* bih[2] = {(const float*)d_in[4], (const float*)d_in[8]};
  const float* bhh[2] = {(const float*)d_in[5], (const float*)d_in[9]};
  const float* fuse_w = (const float*)d_in[10];
  const float* fuse_b = (const float*)d_in[11];
  const float* c1w = (const float*)d_in[12];
  const float* bn1[4] = {(const float*)d_in[13], (const float*)d_in[14],
                         (const float*)d_in[15], (const float*)d_in[16]};
  const float* c2w = (const float*)d_in[17];
  const float* bn2[4] = {(const float*)d_in[18], (const float*)d_in[19],
                         (const float*)d_in[20], (const float*)d_in[21]};
  const float* c3w = (const float*)d_in[22];
  const float* bn3[4] = {(const float*)d_in[23], (const float*)d_in[24],
                         (const float*)d_in[25], (const float*)d_in[26]};
  const float* c4w = (const float*)d_in[27];
  const float* bn4[4] = {(const float*)d_in[28], (const float*)d_in[29],
                         (const float*)d_in[30], (const float*)d_in[31]};
  const float* c5w = (const float*)d_in[32];
  const float* c5b = (const float*)d_in[33];
  const int* proj = (const int*)d_in[34];
  const unsigned char* mask = (const unsigned char*)d_in[35];
  float* out = (float*)d_out;

  // ---- carve workspace ----
  char* wp = (char*)d_ws;
  auto carve = [&](size_t bytes) { void* p = wp; wp += (bytes + 255) & ~(size_t)255; return p; };
  float*    state   = (float*)carve((size_t)MSIZE * MEM_C_ * 4);
  float*    state_r = (float*)carve((size_t)MSIZE * MEM_C_ * 4);
  float*    hm      = (float*)carve((size_t)MSIZE * 4);
  float*    hm_r    = (float*)carve((size_t)MSIZE * 4);
  unsigned* keyb    = (unsigned*)carve((size_t)MSIZE * 4);
  int*      argb    = (int*)carve((size_t)MSIZE * 4);
  unsigned* obs     = (unsigned*)carve((size_t)MSIZE * 4);
  _Float16* Gh      = (_Float16*)carve((size_t)MSIZE * EGO_C * 2);
  _Float16* ShD[2]  = {(_Float16*)carve((size_t)MSIZE * MEM_C_ * 2),
                       (_Float16*)carve((size_t)MSIZE * MEM_C_ * 2)};
  float*    gi      = (float*)carve((size_t)MSIZE * GATE3 * 4);   // 61.4 MB
  float*    gh      = (float*)carve((size_t)MSIZE * GATE3 * 4);   // 61.4 MB
  _Float16* WihH[2] = {(_Float16*)carve((size_t)GATE3 * EGO_C * 2),
                       (_Float16*)carve((size_t)GATE3 * EGO_C * 2)};
  _Float16* WhhH[2] = {(_Float16*)carve((size_t)GATE3 * MEM_C_ * 2),
                       (_Float16*)carve((size_t)GATE3 * MEM_C_ * 2)};
  _Float16* wfuseH = (_Float16*)carve((size_t)128 * 256 * 2);
  _Float16* w1H = (_Float16*)carve((size_t)128 * 49 * 128 * 2);
  _Float16* w2H = (_Float16*)carve((size_t)64 * 9 * 128 * 2);
  _Float16* w3H = (_Float16*)carve((size_t)48 * 9 * 64 * 2);
  _Float16* w4H = (_Float16*)carve((size_t)48 * 9 * 64 * 2);
  _Float16* w5H = (_Float16*)carve((size_t)32 * 64 * 2);
  // post-scan buffers aliased into gi/gh (free after the scan)
  _Float16* Xfuse = (_Float16*)gi;                                  // 20.5 MB
  _Float16* actA  = (_Float16*)((char*)gi + 24u * 1024 * 1024);     // 10.2 MB
  _Float16* actB  = (_Float16*)((char*)gi + 40u * 1024 * 1024);     // 10.2 MB
  float*    craw  = gh;                                             // 20.5 MB

  const int TB = 256;

  // ---- pack weights (f32 -> f16, channel-padded) ----
  for (int d = 0; d < 2; ++d) {
    k_f32_to_f16<<<cdiv(GATE3 * EGO_C, TB), TB, 0, stream>>>(Wih[d], WihH[d], GATE3 * EGO_C);
    k_f32_to_f16<<<cdiv(GATE3 * MEM_C_, TB), TB, 0, stream>>>(Whh[d], WhhH[d], GATE3 * MEM_C_);
  }
  k_pack_conv<<<cdiv(128 * 256, TB), TB, 0, stream>>>(fuse_w, wfuseH, 128, 128, 256, 256, 1, 1);
  k_pack_conv<<<cdiv(128 * 49 * 128, TB), TB, 0, stream>>>(c1w, w1H, 128, 128, 128, 128, 7, 7);
  k_pack_conv<<<cdiv(64 * 9 * 128, TB), TB, 0, stream>>>(c2w, w2H, 64, 64, 128, 128, 3, 3);
  k_pack_conv<<<cdiv(48 * 9 * 64, TB), TB, 0, stream>>>(c3w, w3H, 48, 48, 64, 64, 3, 3);
  k_pack_conv<<<cdiv(48 * 9 * 64, TB), TB, 0, stream>>>(c4w, w4H, 48, 48, 48, 64, 3, 3);
  k_pack_conv<<<cdiv(32 * 64, TB), TB, 0, stream>>>(c5w, w5H, 21, 32, 48, 64, 1, 1);

  // ---- init persistent scan state ----
  k_zero32<<<cdiv(MSIZE * MEM_C_, TB), TB, 0, stream>>>((unsigned*)state, MSIZE * MEM_C_);
  k_zero32<<<cdiv(MSIZE * MEM_C_, TB), TB, 0, stream>>>((unsigned*)state_r, MSIZE * MEM_C_);
  k_zero32<<<cdiv(MSIZE * MEM_C_ / 2, TB), TB, 0, stream>>>((unsigned*)ShD[0], MSIZE * MEM_C_ / 2);
  k_zero32<<<cdiv(MSIZE * MEM_C_ / 2, TB), TB, 0, stream>>>((unsigned*)ShD[1], MSIZE * MEM_C_ / 2);
  k_zero32<<<cdiv(MSIZE, TB), TB, 0, stream>>>((unsigned*)hm, MSIZE);
  k_zero32<<<cdiv(MSIZE, TB), TB, 0, stream>>>((unsigned*)hm_r, MSIZE);
  k_zero32<<<cdiv(MSIZE, TB), TB, 0, stream>>>(obs, MSIZE);

  // ---- temporal scan: forward + reverse direction per step ----
  for (int t = 0; t < T_STEPS; ++t) {
    for (int d = 0; d < 2; ++d) {
      int ts = (d == 0) ? t : (T_STEPS - 1 - t);
      const float* feat = features + (size_t)ts * EGO_C * FH_ * FW_;
      const int* pr = proj + (size_t)ts * NPIX * 2;
      const unsigned char* mk = mask + (size_t)ts * NPIX;
      const float* hg = heights + (size_t)ts * NPIX;
      float* st = (d == 0) ? state : state_r;
      float* hmap = (d == 0) ? hm : hm_r;

      k_cell_prep<<<cdiv(MSIZE, TB), TB, 0, stream>>>(hmap, keyb, argb);
      k_scatter_max<<<cdiv(NPIX, TB), TB, 0, stream>>>(pr, mk, hg, keyb);
      k_scatter_arg<<<cdiv(NPIX, TB), TB, 0, stream>>>(pr, mk, hg, hmap, keyb, argb);
      k_finalize_gather<<<MSIZE / 4, 256, 0, stream>>>(keyb, argb, hmap, obs, d == 0 ? 1 : 0,
                                                       feat, Gh);
      // gi = G @ Wih^T  (M=40000, K=64,  N=384) ; gh = S @ Whh^T (K=128)
      launch_conv<4, 2, MAPH, MAPW, EGO_C, 1, 1, 0, GATE3>(Gh, WihH[d], gi, stream);
      launch_conv<4, 2, MAPH, MAPW, MEM_C_, 1, 1, 0, GATE3>(ShD[d], WhhH[d], gh, stream);
      k_gru_point<<<cdiv(MSIZE * MEM_C_, TB), TB, 0, stream>>>(gi, gh, bih[d], bhh[d], argb,
                                                               st, ShD[d]);
    }
  }

  // ---- outputs: obs + hm (forward direction) ----
  k_out_obs_hm<<<cdiv(MSIZE, TB), TB, 0, stream>>>(obs, hm, out);

  // ---- decoder head (all GEMMs via WMMA) ----
  k_concat_half<<<cdiv(MSIZE * 256, TB), TB, 0, stream>>>(state, state_r, Xfuse);
  launch_conv<4, 2, MAPH, MAPW, 256, 1, 1, 0, 128>(Xfuse, wfuseH, craw, stream);
  k_bias_half<<<cdiv(MSIZE * 128, TB), TB, 0, stream>>>(craw, fuse_b, actA, 128);
  launch_conv<4, 2, MAPH, MAPW, 128, 7, 7, 3, 128>(actA, w1H, craw, stream);
  k_bnrelu<<<cdiv(MSIZE * 128, TB), TB, 0, stream>>>(craw, 128, actB, 128,
                                                     bn1[0], bn1[1], bn1[2], bn1[3]);
  launch_conv<4, 2, MAPH, MAPW, 128, 3, 3, 1, 64>(actB, w2H, craw, stream);
  k_bnrelu<<<cdiv(MSIZE * 64, TB), TB, 0, stream>>>(craw, 64, actA, 64,
                                                    bn2[0], bn2[1], bn2[2], bn2[3]);
  launch_conv<3, 2, MAPH, MAPW, 64, 3, 3, 1, 48>(actA, w3H, craw, stream);
  k_bnrelu<<<cdiv(MSIZE * 64, TB), TB, 0, stream>>>(craw, 48, actB, 64,
                                                    bn3[0], bn3[1], bn3[2], bn3[3]);
  launch_conv<3, 2, MAPH, MAPW, 64, 3, 3, 1, 48>(actB, w4H, craw, stream);
  k_bnrelu<<<cdiv(MSIZE * 64, TB), TB, 0, stream>>>(craw, 48, actA, 64,
                                                    bn4[0], bn4[1], bn4[2], bn4[3]);
  launch_conv<2, 2, MAPH, MAPW, 64, 1, 1, 0, 32>(actA, w5H, craw, stream);
  k_c5_out<<<cdiv(SEM_N * MSIZE, TB), TB, 0, stream>>>(craw, c5b, out);
}
